// WeightOnlyInt8Linear_26353919328818
// MI455X (gfx1250) — compile-verified, matrix operand order checked by reading, not run on hardware
//
#include <hip/hip_runtime.h>
#include <cstdint>

typedef __attribute__((ext_vector_type(16))) _Float16 v16h;
typedef __attribute__((ext_vector_type(8)))  _Float16 v8h;
typedef __attribute__((ext_vector_type(4)))  _Float16 v4h;
typedef __attribute__((ext_vector_type(8)))  float    v8f;
typedef __attribute__((ext_vector_type(4)))  unsigned v4u;
typedef __attribute__((ext_vector_type(4)))  int      v4i;
typedef __attribute__((ext_vector_type(8)))  int      v8i;

constexpr int Mdim = 4 * 2048;   // B*S
constexpr int Ndim = 11008;      // DOUT
constexpr int Kdim = 4096;       // DIN

// ---------------- TDM availability probe (device pass) ----------------
#if defined(__has_builtin)
#if __has_builtin(__builtin_amdgcn_tensor_load_to_lds) && \
    __has_builtin(__builtin_amdgcn_s_wait_tensorcnt)
#define HAVE_TDM 1
#endif
#endif

// =====================================================================
// Pre-conversion kernels: one-time f32->f16 and i32->f16 (4 elems/thread)
// =====================================================================
__global__ __launch_bounds__(256)
void cvt_f32_to_f16(const float* __restrict__ in, _Float16* __restrict__ out, long n4) {
  long i = (long)blockIdx.x * blockDim.x + threadIdx.x;
  if (i < n4) {
    float4 v = ((const float4*)in)[i];
    v4h h; h[0] = (_Float16)v.x; h[1] = (_Float16)v.y;
           h[2] = (_Float16)v.z; h[3] = (_Float16)v.w;
    ((v4h*)out)[i] = h;
  }
}

__global__ __launch_bounds__(256)
void cvt_i32_to_f16(const int* __restrict__ in, _Float16* __restrict__ out, long n4) {
  long i = (long)blockIdx.x * blockDim.x + threadIdx.x;
  if (i < n4) {
    int4 v = ((const int4*)in)[i];
    v4h h; h[0] = (_Float16)v.x; h[1] = (_Float16)v.y;
           h[2] = (_Float16)v.z; h[3] = (_Float16)v.w;
    ((v4h*)out)[i] = h;
  }
}

// =====================================================================
// Main GEMM on pre-converted f16: 128x256 block, 8 waves, 64x64/wave,
// TDM (tensor_load_to_lds) double-buffered staging with HW row padding.
// =====================================================================
constexpr int BM2  = 128;
constexpr int BN2  = 256;
constexpr int BK2  = 32;
constexpr int LDK2 = BK2 + 8;            // 40 halves = 80B row stride (bank-conflict-free b128)
constexpr int KT2  = Kdim / BK2;         // 128 k-steps

// LDS layout (in halves)
constexpr int AS0 = 0;
constexpr int AS1 = BM2 * LDK2;          // 5120
constexpr int BS0 = 2 * BM2 * LDK2;      // 10240
constexpr int BS1 = BS0 + BN2 * LDK2;    // 20480
constexpr int SMEM_HALVES = BS1 + BN2 * LDK2;         // 30720
constexpr size_t SMEM_BYTES = (size_t)SMEM_HALVES * 2; // 61440 B

#if defined(HAVE_TDM)
// Issue one 2D TDM tile load: tile_rows x 32 halves, tensor row stride Kdim halves.
// pad_enable inserts 4 dwords (8 halves) after every 16 dwords (32 halves) of data,
// reproducing the LDK2=40 padded LDS row stride in hardware.
__device__ __forceinline__ void tdm_load_tile(unsigned lds_byte,
                                              const _Float16* gptr,
                                              unsigned tile_rows) {
  unsigned long long ga = (unsigned long long)(uintptr_t)gptr;
  v4u g0;
  g0[0] = 1u;                                              // count=1, user descriptor
  g0[1] = lds_byte;                                        // lds_addr (bytes)
  g0[2] = (unsigned)ga;                                    // global_addr[31:0]
  g0[3] = ((unsigned)(ga >> 32) & 0x01FFFFFFu) | (2u << 30); // addr[56:32] | type=2
  v8i g1;
  g1[0] = (int)((1u << 16)        // data_size = 1 (2-byte elements)
              | (1u << 20)        // pad_enable
              | (3u << 22)        // pad_interval: 16 dwords of data
              | (3u << 25));      // pad_amount: 4 dwords of pad
  g1[1] = (int)(((unsigned)Kdim & 0xFFFFu) << 16);         // tensor_dim0[15:0]
  g1[2] = (int)((((unsigned)Kdim) >> 16)                   // tensor_dim0[31:16]
              | ((tile_rows & 0xFFFFu) << 16));            // tensor_dim1[15:0] (tile fits; no OOB)
  g1[3] = (int)((tile_rows >> 16) | (32u << 16));          // tensor_dim1 hi | tile_dim0=32
  g1[4] = (int)tile_rows;                                  // tile_dim1 (tile_dim2=0)
  g1[5] = (int)Kdim;                                       // tensor_dim0_stride low32 (elements)
  g1[6] = 0;
  g1[7] = 0;
  v4i z4 = {0, 0, 0, 0};
#if __clang_major__ >= 23
  v8i z8 = {0, 0, 0, 0, 0, 0, 0, 0};
  __builtin_amdgcn_tensor_load_to_lds(g0, g1, z4, z4, z8, 0);
#else
  __builtin_amdgcn_tensor_load_to_lds(g0, g1, z4, z4, 0);
#endif
}
#endif

__global__ __launch_bounds__(256)
void wmma_gemm_f16(const _Float16* __restrict__ Ah,
                   const _Float16* __restrict__ Bh,
                   const float* __restrict__ scales,
                   float* __restrict__ out) {
  extern __shared__ __align__(16) _Float16 sm[];

  const int tid   = threadIdx.x;
  const int lane  = tid & 31;
  const int wave  = tid >> 5;     // 0..7
  const int waveM = wave & 1;     // 2 waves along M (64 rows each)
  const int waveN = wave >> 1;    // 4 waves along N (64 cols each)
  const int l16   = lane & 15;
  const int half  = lane >> 4;

  const int nBlk = blockIdx.x * BN2;
  const int mBlk = blockIdx.y * BM2;

  const int asOff[2] = {AS0, AS1};
  const int bsOff[2] = {BS0, BS1};

  const v8f vzero = {};
  v8f acc[4][4];
#pragma unroll
  for (int i = 0; i < 4; ++i)
#pragma unroll
    for (int j = 0; j < 4; ++j)
      acc[i][j] = vzero;

  // fragment loads (2x ds_load_b128 each) + 16 WMMAs per k-step
  auto compute = [&](int cur) {
    const _Float16* asb = sm + asOff[cur];
    const _Float16* bsb = sm + bsOff[cur];
    v16h afrag[4];
#pragma unroll
    for (int ti = 0; ti < 4; ++ti) {
      const _Float16* ap = asb + (waveM * 64 + ti * 16 + l16) * LDK2 + half * 8;
      v8h lo = *(const v8h*)ap;
      v8h hi = *(const v8h*)(ap + 16);
      afrag[ti] = __builtin_shufflevector(lo, hi, 0,1,2,3,4,5,6,7,8,9,10,11,12,13,14,15);
    }
#pragma unroll
    for (int tj = 0; tj < 4; ++tj) {
      const _Float16* bp = bsb + (waveN * 64 + tj * 16 + l16) * LDK2 + half * 8;
      v8h lo = *(const v8h*)bp;
      v8h hi = *(const v8h*)(bp + 16);
      v16h bfrag = __builtin_shufflevector(lo, hi, 0,1,2,3,4,5,6,7,8,9,10,11,12,13,14,15);
#pragma unroll
      for (int ti = 0; ti < 4; ++ti) {
        acc[ti][tj] = __builtin_amdgcn_wmma_f32_16x16x32_f16(
            false, afrag[ti], false, bfrag, (short)0, acc[ti][tj], false, false);
      }
    }
  };

#if defined(HAVE_TDM)
  // ---- TDM double-buffered pipeline: wave 0 drives the DMA ----
  const unsigned ldsBase = __builtin_amdgcn_groupstaticsize();
  if (wave == 0) {
    tdm_load_tile(ldsBase + (unsigned)AS0 * 2, Ah + (size_t)mBlk * Kdim, BM2);
    tdm_load_tile(ldsBase + (unsigned)BS0 * 2, Bh + (size_t)nBlk * Kdim, BN2);
  }
  for (int kt = 0; kt < KT2; ++kt) {
    const int cur = kt & 1;
    if (wave == 0) {
      if (kt + 1 < KT2) {
        const int k0 = (kt + 1) * BK2;
        tdm_load_tile(ldsBase + (unsigned)asOff[cur ^ 1] * 2,
                      Ah + (size_t)mBlk * Kdim + k0, BM2);
        tdm_load_tile(ldsBase + (unsigned)bsOff[cur ^ 1] * 2,
                      Bh + (size_t)nBlk * Kdim + k0, BN2);
        __builtin_amdgcn_s_wait_tensorcnt(2);  // tile kt's pair complete (in-order)
      } else {
        __builtin_amdgcn_s_wait_tensorcnt(0);
      }
    }
    __syncthreads();        // tile cur visible to all waves
    compute(cur);
    __syncthreads();        // all reads of buf cur done before it is re-filled
  }
#else
  // ---- Fallback: register-staged double buffering (pure f16 b128 copies) ----
  const int srow = tid >> 2;       // 0..63
  const int sch  = tid & 3;        // 16B chunk within a 64B row
  {
#pragma unroll
    for (int p = 0; p < 2; ++p) {
      const int row = p * 64 + srow;
      *(v8h*)&sm[AS0 + row * LDK2 + sch * 8] =
          *(const v8h*)(Ah + (size_t)(mBlk + row) * Kdim + sch * 8);
    }
#pragma unroll
    for (int p = 0; p < 4; ++p) {
      const int row = p * 64 + srow;
      *(v8h*)&sm[BS0 + row * LDK2 + sch * 8] =
          *(const v8h*)(Bh + (size_t)(nBlk + row) * Kdim + sch * 8);
    }
  }
  __syncthreads();
  for (int kt = 0; kt < KT2; ++kt) {
    const int cur = kt & 1;
    v8h aR[2], bR[4];
    if (kt + 1 < KT2) {
      const int k0 = (kt + 1) * BK2;
#pragma unroll
      for (int p = 0; p < 2; ++p)
        aR[p] = *(const v8h*)(Ah + (size_t)(mBlk + p * 64 + srow) * Kdim + k0 + sch * 8);
#pragma unroll
      for (int p = 0; p < 4; ++p)
        bR[p] = *(const v8h*)(Bh + (size_t)(nBlk + p * 64 + srow) * Kdim + k0 + sch * 8);
    }
    compute(cur);
    if (kt + 1 < KT2) {
#pragma unroll
      for (int p = 0; p < 2; ++p)
        *(v8h*)&sm[asOff[cur ^ 1] + (p * 64 + srow) * LDK2 + sch * 8] = aR[p];
#pragma unroll
      for (int p = 0; p < 4; ++p)
        *(v8h*)&sm[bsOff[cur ^ 1] + (p * 64 + srow) * LDK2 + sch * 8] = bR[p];
    }
    __syncthreads();
  }
#endif

  // epilogue: per-channel scale. C/D layout: VGPR r -> (M = r + 8*half, N = l16)
#pragma unroll
  for (int tj = 0; tj < 4; ++tj) {
    const int n = nBlk + waveN * 64 + tj * 16 + l16;
    const float s = scales[n];
#pragma unroll
    for (int ti = 0; ti < 4; ++ti) {
      const int m0 = mBlk + waveM * 64 + ti * 16 + half * 8;
      float* op = out + (size_t)m0 * Ndim + n;
      v8f c = acc[ti][tj];
#pragma unroll
      for (int r = 0; r < 8; ++r)
        op[(size_t)r * Ndim] = c[r] * s;
    }
  }
}

// =====================================================================
// Fallback GEMM (round-1 kernel): in-loop conversion, used if ws too small
// =====================================================================
constexpr int BM  = 128;
constexpr int BN  = 128;
constexpr int BK  = 32;
constexpr int LDK = BK + 8;
constexpr int KT  = Kdim / BK;

__global__ __launch_bounds__(256)
void w8a16_wmma_gemm(const float* __restrict__ x,
                     const int*   __restrict__ w,
                     const float* __restrict__ scales,
                     float*       __restrict__ out) {
  __shared__ __align__(16) _Float16 As[2][BM * LDK];
  __shared__ __align__(16) _Float16 Bs[2][BN * LDK];

  const int tid   = threadIdx.x;
  const int lane  = tid & 31;
  const int wave  = tid >> 5;
  const int waveM = wave & 1;
  const int waveN = wave >> 1;
  const int l16   = lane & 15;
  const int half  = lane >> 4;

  const int nBlk = blockIdx.x * BN;
  const int mBlk = blockIdx.y * BM;

  const int srow = tid >> 3;
  const int sq   = tid & 7;

  const v8f vzero = {};
  v8f acc[4][2];
#pragma unroll
  for (int i = 0; i < 4; ++i)
#pragma unroll
    for (int j = 0; j < 2; ++j)
      acc[i][j] = vzero;

#pragma unroll
  for (int p = 0; p < 4; ++p) {
    const int row = p * 32 + srow;
    const float4 av = *(const float4*)(x + (size_t)(mBlk + row) * Kdim + sq * 4);
    v4h ha; ha[0] = (_Float16)av.x; ha[1] = (_Float16)av.y;
            ha[2] = (_Float16)av.z; ha[3] = (_Float16)av.w;
    *(v4h*)&As[0][row * LDK + sq * 4] = ha;
    const int4 bv = *(const int4*)(w + (size_t)(nBlk + row) * Kdim + sq * 4);
    v4h hb; hb[0] = (_Float16)bv.x; hb[1] = (_Float16)bv.y;
            hb[2] = (_Float16)bv.z; hb[3] = (_Float16)bv.w;
    *(v4h*)&Bs[0][row * LDK + sq * 4] = hb;
  }
  __syncthreads();

  for (int kt = 0; kt < KT; ++kt) {
    const int cur = kt & 1;
    float4 areg[4];
    int4   breg[4];
    if (kt + 1 < KT) {
      const int k0 = (kt + 1) * BK;
#pragma unroll
      for (int p = 0; p < 4; ++p) {
        const int row = p * 32 + srow;
        areg[p] = *(const float4*)(x + (size_t)(mBlk + row) * Kdim + k0 + sq * 4);
        breg[p] = *(const int4*)(w + (size_t)(nBlk + row) * Kdim + k0 + sq * 4);
      }
    }
    v16h afrag[4];
#pragma unroll
    for (int ti = 0; ti < 4; ++ti) {
      const _Float16* ap = &As[cur][(waveM * 64 + ti * 16 + l16) * LDK + half * 8];
      v8h lo = *(const v8h*)ap;
      v8h hi = *(const v8h*)(ap + 16);
      afrag[ti] = __builtin_shufflevector(lo, hi, 0,1,2,3,4,5,6,7,8,9,10,11,12,13,14,15);
    }
#pragma unroll
    for (int tj = 0; tj < 2; ++tj) {
      const _Float16* bp = &Bs[cur][(waveN * 32 + tj * 16 + l16) * LDK + half * 8];
      v8h lo = *(const v8h*)bp;
      v8h hi = *(const v8h*)(bp + 16);
      v16h bfrag = __builtin_shufflevector(lo, hi, 0,1,2,3,4,5,6,7,8,9,10,11,12,13,14,15);
#pragma unroll
      for (int ti = 0; ti < 4; ++ti) {
        acc[ti][tj] = __builtin_amdgcn_wmma_f32_16x16x32_f16(
            false, afrag[ti], false, bfrag, (short)0, acc[ti][tj], false, false);
      }
    }
    if (kt + 1 < KT) {
#pragma unroll
      for (int p = 0; p < 4; ++p) {
        const int row = p * 32 + srow;
        v4h ha; ha[0] = (_Float16)areg[p].x; ha[1] = (_Float16)areg[p].y;
                ha[2] = (_Float16)areg[p].z; ha[3] = (_Float16)areg[p].w;
        *(v4h*)&As[cur ^ 1][row * LDK + sq * 4] = ha;
        v4h hb; hb[0] = (_Float16)breg[p].x; hb[1] = (_Float16)breg[p].y;
                hb[2] = (_Float16)breg[p].z; hb[3] = (_Float16)breg[p].w;
        *(v4h*)&Bs[cur ^ 1][row * LDK + sq * 4] = hb;
      }
    }
    __syncthreads();
  }

#pragma unroll
  for (int tj = 0; tj < 2; ++tj) {
    const int n = nBlk + waveN * 32 + tj * 16 + l16;
    const float s = scales[n];
#pragma unroll
    for (int ti = 0; ti < 4; ++ti) {
      const int m0 = mBlk + waveM * 64 + ti * 16 + half * 8;
      float* op = out + (size_t)m0 * Ndim + n;
      v8f c = acc[ti][tj];
#pragma unroll
      for (int r = 0; r < 8; ++r)
        op[(size_t)r * Ndim] = c[r] * s;
    }
  }
}

// =====================================================================
extern "C" void kernel_launch(void* const* d_in, const int* in_sizes, int n_in,
                              void* d_out, int out_size, void* d_ws, size_t ws_size,
                              hipStream_t stream) {
  const float* x      = (const float*)d_in[0];
  const int*   w      = (const int*)d_in[1];
  const float* scales = (const float*)d_in[2];
  float*       out    = (float*)d_out;

  const size_t wsA = (size_t)Mdim * Kdim * sizeof(_Float16);   // 67 MB
  const size_t wsB = (size_t)Ndim * Kdim * sizeof(_Float16);   // 90 MB

  if (ws_size >= wsA + wsB) {
    _Float16* Ah = (_Float16*)d_ws;
    _Float16* Bh = (_Float16*)((char*)d_ws + wsA);
    const long n4a = (long)Mdim * Kdim / 4;
    const long n4b = (long)Ndim * Kdim / 4;
    cvt_f32_to_f16<<<dim3((unsigned)((n4a + 255) / 256)), 256, 0, stream>>>(x, Ah, n4a);
    cvt_i32_to_f16<<<dim3((unsigned)((n4b + 255) / 256)), 256, 0, stream>>>(w, Bh, n4b);
    dim3 grid(Ndim / BN2, Mdim / BM2);   // x over N: consecutive WGs share the A strip in L2
    wmma_gemm_f16<<<grid, 256, SMEM_BYTES, stream>>>(Ah, Bh, scales, out);
  } else {
    dim3 grid(Ndim / BN, Mdim / BM);
    w8a16_wmma_gemm<<<grid, 256, 0, stream>>>(x, w, scales, out);
  }
}
